// Q_bbox_embed_12876311953738
// MI455X (gfx1250) — 8-bit matrix operand order repaired and checked by reading, compiled, not run on hardware
//
#include <hip/hip_runtime.h>
#include <stdint.h>


typedef __attribute__((ext_vector_type(8))) int v8i;

#define QMLP_N   (6 * 64 * 300)
#define QMLP_D   256
#define QMLP_EPS 1e-8f

#define OFF_SCALE 0
#define OFF_WS1   256
#define OFF_WS2   1280
#define OFF_WS3   2304
#define OFF_W1Q   4096
#define OFF_W2Q   69632
#define OFF_W3Q   135168
#define OFF_AI8   139264
#define OFF_AF32  29630464UL

__device__ __forceinline__ float wave_max32(float m) {
  #pragma unroll
  for (int off = 16; off > 0; off >>= 1)
    m = fmaxf(m, __shfl_xor(m, off, 32));
  return m;
}

__device__ __forceinline__ int quant_clip(float x, float s) {
  float r = rintf(x / s);
  r = fminf(fmaxf(r, -128.0f), 127.0f);
  return (int)r;
}

__global__ void zero_scales(uint32_t* p) {
  if (threadIdx.x < 4) p[threadIdx.x] = 0u;
}

__global__ __launch_bounds__(256) void absmax_f32(const float4* __restrict__ x,
                                                  size_t n4,
                                                  uint32_t* __restrict__ acc) {
  float m = 0.0f;
  size_t stride = (size_t)gridDim.x * blockDim.x;
  for (size_t i = (size_t)blockIdx.x * blockDim.x + threadIdx.x; i < n4; i += stride) {
    float4 v = x[i];
    m = fmaxf(m, fmaxf(fmaxf(fabsf(v.x), fabsf(v.y)), fmaxf(fabsf(v.z), fabsf(v.w))));
  }
  m = wave_max32(m);
  if ((threadIdx.x & 31) == 0) atomicMax(acc, __float_as_uint(m));
}

__global__ __launch_bounds__(256) void wquant(const float* __restrict__ w,
                                              int validRows,
                                              int8_t* __restrict__ wq,
                                              float* __restrict__ wsc) {
  __shared__ float red[8];
  __shared__ float s_scale;
  const int r = blockIdx.x;
  const int t = threadIdx.x;
  const float v = (r < validRows) ? w[(size_t)r * QMLP_D + t] : 0.0f;
  float m = wave_max32(fabsf(v));
  if ((t & 31) == 0) red[t >> 5] = m;
  __syncthreads();
  if (t == 0) {
    float mm = red[0];
    #pragma unroll
    for (int i = 1; i < 8; ++i) mm = fmaxf(mm, red[i]);
    s_scale = fmaxf(mm, QMLP_EPS) * (1.0f / 127.0f);
  }
  __syncthreads();
  const float sc = s_scale;
  wq[(size_t)r * QMLP_D + t] = (int8_t)quant_clip(v, sc);
  if (t == 0) wsc[r] = sc;
}

__global__ __launch_bounds__(256) void quant_act_k(const float4* __restrict__ x,
                                                   uint32_t* __restrict__ q,
                                                   const uint32_t* __restrict__ amaxBits,
                                                   size_t n4) {
  const float s = fmaxf(__uint_as_float(*amaxBits), QMLP_EPS) * (1.0f / 127.0f);
  size_t stride = (size_t)gridDim.x * blockDim.x;
  for (size_t i = (size_t)blockIdx.x * blockDim.x + threadIdx.x; i < n4; i += stride) {
    float4 v = x[i];
    uint32_t p = ((uint32_t)(quant_clip(v.x, s) & 0xff)) |
                 ((uint32_t)(quant_clip(v.y, s) & 0xff) << 8) |
                 ((uint32_t)(quant_clip(v.z, s) & 0xff) << 16) |
                 ((uint32_t)(quant_clip(v.w, s) & 0xff) << 24);
    q[i] = p;
  }
}

__global__ __launch_bounds__(256) void gemm_i8_relu(const int8_t* __restrict__ A,
                                                    const uint32_t* __restrict__ amaxBits,
                                                    const int8_t* __restrict__ W,
                                                    const float* __restrict__ wscale,
                                                    const float* __restrict__ bias,
                                                    float* __restrict__ out,
                                                    uint32_t* __restrict__ outMaxBits) {
  const int tid  = threadIdx.x;
  const int wave = tid >> 5;
  const int lane = tid & 31;
  const int lrow = lane & 15;
  const int half = lane >> 4;
  const int mw   = wave >> 2;
  const int nw   = wave & 3;
  const int M0   = blockIdx.x * 32 + mw * 16;
  const int row  = M0 + lrow;

  v8i accs[4] = {{}, {}, {}, {}};

  #pragma unroll
  for (int ks = 0; ks < 4; ++ks) {
    const int kb = ks * 64;
    v8i a;
    const int8_t* ap = A + (size_t)row * QMLP_D + kb + half * 16;
    #pragma unroll
    for (int j = 0; j < 4; ++j) {
      uint64_t qw = *reinterpret_cast<const uint64_t*>(ap + (j >> 1) * 32 + (j & 1) * 8);
      a[2 * j]     = (int)(uint32_t)(qw & 0xffffffffu);
      a[2 * j + 1] = (int)(uint32_t)(qw >> 32);
    }
    #pragma unroll
    for (int nt = 0; nt < 4; ++nt) {
      const int o = nw * 64 + nt * 16 + lrow;
      const int8_t* bp = W + (size_t)o * QMLP_D + kb + half * 16;
      const int4 lo = *reinterpret_cast<const int4*>(bp);
      const int4 hi = *reinterpret_cast<const int4*>(bp + 32);
      v8i b;
      b[0] = lo.x; b[1] = lo.y; b[2] = lo.z; b[3] = lo.w;
      b[4] = hi.x; b[5] = hi.y; b[6] = hi.z; b[7] = hi.w;
      accs[nt] = __builtin_amdgcn_wmma_i32_16x16x64_iu8(
           true, a,  true, b, accs[nt],
           false,  false);
    }
  }

  const float s_act = fmaxf(__uint_as_float(*amaxBits), QMLP_EPS) * (1.0f / 127.0f);
  float lmax = 0.0f;
  #pragma unroll
  for (int nt = 0; nt < 4; ++nt) {
    const int o   = nw * 64 + nt * 16 + lrow;
    const float bs   = wscale[o] * s_act;
    const float bint = rintf(bias[o] / bs);
    #pragma unroll
    for (int d = 0; d < 8; ++d) {
      float v = ((float)accs[nt][d] + bint) * bs;
      v = fmaxf(v, 0.0f);
      out[(size_t)(M0 + d + half * 8) * QMLP_D + o] = v;
      lmax = fmaxf(lmax, v);
    }
  }
  lmax = wave_max32(lmax);
  if (lane == 0) atomicMax(outMaxBits, __float_as_uint(lmax));
}

__global__ __launch_bounds__(256) void gemm_i8_final(const int8_t* __restrict__ A,
                                                     const uint32_t* __restrict__ amaxBits,
                                                     const int8_t* __restrict__ W,
                                                     const float* __restrict__ wscale,
                                                     const float* __restrict__ bias,
                                                     float* __restrict__ out) {
  const int tid  = threadIdx.x;
  const int wave = tid >> 5;
  const int lane = tid & 31;
  const int lrow = lane & 15;
  const int half = lane >> 4;
  const int M0   = blockIdx.x * 128 + wave * 16;
  const int row  = M0 + lrow;

  v8i acc = {};
  #pragma unroll
  for (int ks = 0; ks < 4; ++ks) {
    const int kb = ks * 64;
    v8i a;
    const int8_t* ap = A + (size_t)row * QMLP_D + kb + half * 16;
    #pragma unroll
    for (int j = 0; j < 4; ++j) {
      uint64_t qw = *reinterpret_cast<const uint64_t*>(ap + (j >> 1) * 32 + (j & 1) * 8);
      a[2 * j]     = (int)(uint32_t)(qw & 0xffffffffu);
      a[2 * j + 1] = (int)(uint32_t)(qw >> 32);
    }
    const int8_t* bp = W + (size_t)lrow * QMLP_D + kb + half * 16;
    const int4 lo = *reinterpret_cast<const int4*>(bp);
    const int4 hi = *reinterpret_cast<const int4*>(bp + 32);
    v8i b;
    b[0] = lo.x; b[1] = lo.y; b[2] = lo.z; b[3] = lo.w;
    b[4] = hi.x; b[5] = hi.y; b[6] = hi.z; b[7] = hi.w;
    acc = __builtin_amdgcn_wmma_i32_16x16x64_iu8(true, a, true, b, acc, false, false);
  }

  if (lrow < 4) {
    const float s_act = fmaxf(__uint_as_float(*amaxBits), QMLP_EPS) * (1.0f / 127.0f);
    const float bs    = wscale[lrow] * s_act;
    const float bint  = rintf(bias[lrow] / bs);
    #pragma unroll
    for (int d = 0; d < 8; ++d)
      out[(size_t)(M0 + d + half * 8) * 4 + lrow] = ((float)acc[d] + bint) * bs;
  }
}

extern "C" void kernel_launch(void* const* d_in, const int* in_sizes, int n_in,
                              void* d_out, int out_size, void* d_ws, size_t ws_size,
                              hipStream_t stream) {
  (void)in_sizes; (void)n_in; (void)out_size; (void)ws_size;
  const float* hs = (const float*)d_in[0];
  const float* w1 = (const float*)d_in[1];
  const float* b1 = (const float*)d_in[2];
  const float* w2 = (const float*)d_in[3];
  const float* b2 = (const float*)d_in[4];
  const float* w3 = (const float*)d_in[5];
  const float* b3 = (const float*)d_in[6];
  float* out = (float*)d_out;

  char* ws = (char*)d_ws;
  uint32_t* scaleAcc = (uint32_t*)(ws + OFF_SCALE);
  float*    ws1      = (float*)(ws + OFF_WS1);
  float*    ws2      = (float*)(ws + OFF_WS2);
  float*    ws3      = (float*)(ws + OFF_WS3);
  int8_t*   w1q      = (int8_t*)(ws + OFF_W1Q);
  int8_t*   w2q      = (int8_t*)(ws + OFF_W2Q);
  int8_t*   w3q      = (int8_t*)(ws + OFF_W3Q);
  int8_t*   ai8      = (int8_t*)(ws + OFF_AI8);
  float*    af32     = (float*)(ws + OFF_AF32);

  const int    N  = QMLP_N;
  const size_t n4 = (size_t)N * QMLP_D / 4;

  zero_scales<<<1, 32, 0, stream>>>(scaleAcc);

  absmax_f32<<<4096, 256, 0, stream>>>((const float4*)hs, n4, scaleAcc + 0);

  wquant<<<256, 256, 0, stream>>>(w1, 256, w1q, ws1);
  wquant<<<256, 256, 0, stream>>>(w2, 256, w2q, ws2);
  wquant<<<16,  256, 0, stream>>>(w3,   4, w3q, ws3);

  quant_act_k<<<8192, 256, 0, stream>>>((const float4*)hs, (uint32_t*)ai8, scaleAcc + 0, n4);

  gemm_i8_relu<<<N / 32, 256, 0, stream>>>(ai8, scaleAcc + 0, w1q, ws1, b1, af32, scaleAcc + 1);

  quant_act_k<<<8192, 256, 0, stream>>>((const float4*)af32, (uint32_t*)ai8, scaleAcc + 1, n4);

  gemm_i8_relu<<<N / 32, 256, 0, stream>>>(ai8, scaleAcc + 1, w2q, ws2, b2, af32, scaleAcc + 2);

  quant_act_k<<<8192, 256, 0, stream>>>((const float4*)af32, (uint32_t*)ai8, scaleAcc + 2, n4);

  gemm_i8_final<<<N / 128, 256, 0, stream>>>(ai8, scaleAcc + 2, w3q, ws3, b3, out);
}
